// InteractionGraphAutoencoder_84490596646922
// MI455X (gfx1250) — compile-verified, matrix operand order checked by reading, not run on hardware
//
#include <hip/hip_runtime.h>
#include <hip/hip_bf16.h>

// ---------------------------------------------------------------------------
// 2-layer GCN (GCNConv x2) for MI455X / gfx1250.
//   h1 = x @ W1                      (bf16 WMMA, f32 accumulate)
//   a1 = Dhat^-1/2 (A+I) Dhat^-1/2 h1 ; h = relu(a1 + b1)
//   h2 = h @ W2                      (bf16 WMMA, f32 accumulate)
//   out = Dhat^-1/2 (A+I) Dhat^-1/2 h2 + b2
// GEMMs: one wave32 per 16xN output slab -> A fragment converted once,
// reused across all N-tiles (x streamed exactly once from HBM).
// Aggregations use hardware global_atomic_add_f32 (feature matrices are
// L2-resident: h1 = 12.8MB, h2 = 6.4MB << 192MB L2).
// ---------------------------------------------------------------------------

typedef __attribute__((ext_vector_type(16))) __bf16 v16bf;
typedef __attribute__((ext_vector_type(8)))  float  v8f;

#define N_FEAT_IN 512
#define N_HID     64
#define N_EMB     32

__device__ __forceinline__ void atomic_add_f32(float* p, float v) {
  unsafeAtomicAdd(p, v);   // global_atomic_add_f32 (no CAS loop)
}

// ---------------- degree / normalization --------------------------------
__global__ void deg_init(float* __restrict__ deg, int n) {
  int i = blockIdx.x * blockDim.x + threadIdx.x;
  if (i < n) deg[i] = 1.0f;              // self loop contributes 1
}

__global__ void deg_scatter(const int* __restrict__ dst, float* __restrict__ deg, int e) {
  int i = blockIdx.x * blockDim.x + threadIdx.x;
  if (i < e) atomic_add_f32(&deg[dst[i]], 1.0f);
}

__global__ void deg_to_rsqrt(float* __restrict__ deg, int n) {
  int i = blockIdx.x * blockDim.x + threadIdx.x;
  if (i < n) deg[i] = rsqrtf(deg[i]);    // deg >= 1 always
}

// ---------------- bf16 WMMA GEMM: C[M,N] = A[M,K] x B[K,N] ---------------
// A row-major f32, B row-major f32, converted to bf16 in registers.
// One wave32 per 16-row slab of C: A fragment loaded once per K-step and
// reused for all N/16 column tiles (N/16 WMMAs per step).
// K % 32 == 0, N % 16 == 0, M % 16 == 0.
template <int K, int N>
__global__ void gemm_bf16_wmma(const float* __restrict__ A,
                               const float* __restrict__ B,
                               float* __restrict__ C, int M) {
  const int wave = (blockIdx.x * blockDim.x + threadIdx.x) >> 5;
  const int lane = threadIdx.x & 31;
  constexpr int NT = N / 16;
  const int mt = wave;                    // one wave per M-tile
  if (mt * 16 >= M) return;               // wave-uniform exit: EXEC all-1 for WMMA

  // ISA 7.12.2 16-bit A layout: lanes 0-15 hold K 0..7 & 16..23 of row M=lane,
  // lanes 16..31 hold K 8..15 & 24..31 of row M=lane-16.
  const int row  = mt * 16 + (lane & 15);        // A row this lane feeds
  const int ncol = lane & 15;                    // column within an N-tile
  const int koff = (lane >> 4) * 8;

  v8f acc[NT] = {};
  for (int k0 = 0; k0 < K; k0 += 32) {
    __builtin_prefetch(&A[(long)row * K + k0 + 64], 0, 0);  // global_prefetch_b8
    v16bf a;
#pragma unroll
    for (int i = 0; i < 8; ++i) {
      a[i]     = (__bf16)A[(long)row * K + k0 + koff + i];
      a[8 + i] = (__bf16)A[(long)row * K + k0 + koff + 16 + i];
    }
#pragma unroll
    for (int nt = 0; nt < NT; ++nt) {
      v16bf b;
#pragma unroll
      for (int i = 0; i < 8; ++i) {
        b[i]     = (__bf16)B[(long)(k0 + koff + i) * N + nt * 16 + ncol];
        b[8 + i] = (__bf16)B[(long)(k0 + koff + 16 + i) * N + nt * 16 + ncol];
      }
      acc[nt] = __builtin_amdgcn_wmma_f32_16x16x32_bf16(
          /*neg_a=*/false, a, /*neg_b=*/false, b,
          /*c_mod=*/(short)0, acc[nt], /*reuse_a=*/false, /*reuse_b=*/false);
    }
  }

  // C/D layout: lanes 0-15 -> M = mt*16 + v, lanes 16-31 -> M = mt*16 + 8 + v
  const int mbase = mt * 16 + ((lane >> 4) ? 8 : 0);
#pragma unroll
  for (int nt = 0; nt < NT; ++nt)
#pragma unroll
    for (int v = 0; v < 8; ++v)
      C[(long)(mbase + v) * N + nt * 16 + ncol] = acc[nt][v];
}

// ---------------- aggregation ------------------------------------------
// out[i,f] = h[i,f] * dinv[i]^2  (self-loop term; seeds the accumulator)
template <int F>
__global__ void selfloop_init(const float* __restrict__ h,
                              const float* __restrict__ dinv,
                              float* __restrict__ out, int n) {
  int idx = blockIdx.x * blockDim.x + threadIdx.x;
  if (idx >= n * F) return;
  int i = idx / F;
  float d = dinv[i];
  out[idx] = h[idx] * d * d;
}

// out[i,f] = h[i,f] * dinv[i]^2 + bias[f]   (layer-2 variant, fuses +b2)
template <int F>
__global__ void selfloop_init_bias(const float* __restrict__ h,
                                   const float* __restrict__ dinv,
                                   const float* __restrict__ bias,
                                   float* __restrict__ out, int n) {
  int idx = blockIdx.x * blockDim.x + threadIdx.x;
  if (idx >= n * F) return;
  int i = idx / F;
  int f = idx & (F - 1);
  float d = dinv[i];
  out[idx] = h[idx] * d * d + bias[f];
}

// out[dst,f] += h[src,f] * dinv[src]*dinv[dst]   (F threads per edge)
template <int F>
__global__ void edge_agg(const int* __restrict__ src, const int* __restrict__ dst,
                         const float* __restrict__ h, const float* __restrict__ dinv,
                         float* __restrict__ out, int E) {
  int t = blockIdx.x * blockDim.x + threadIdx.x;
  int e = t / F;
  int f = t & (F - 1);
  if (e >= E) return;
  int s = src[e];
  int d = dst[e];
  float norm = dinv[s] * dinv[d];
  atomic_add_f32(&out[(long)d * F + f], h[(long)s * F + f] * norm);
}

template <int F>
__global__ void bias_relu(float* __restrict__ h, const float* __restrict__ bias, int total) {
  int idx = blockIdx.x * blockDim.x + threadIdx.x;
  if (idx >= total) return;
  int f = idx & (F - 1);
  h[idx] = fmaxf(h[idx] + bias[f], 0.0f);
}

// ---------------- launch ------------------------------------------------
extern "C" void kernel_launch(void* const* d_in, const int* in_sizes, int n_in,
                              void* d_out, int out_size, void* d_ws, size_t ws_size,
                              hipStream_t stream) {
  const float* x  = (const float*)d_in[0];
  const int*   ei = (const int*)d_in[1];        // [2, E] int32
  const float* W1 = (const float*)d_in[2];
  const float* b1 = (const float*)d_in[3];
  const float* W2 = (const float*)d_in[4];
  const float* b2 = (const float*)d_in[5];
  float* out = (float*)d_out;

  const int N = in_sizes[0] / N_FEAT_IN;        // 50000
  const int E = in_sizes[1] / 2;                // 800000
  const int* src = ei;
  const int* dst = ei + E;

  // workspace layout (floats); total ~32.2 MB
  float* ws   = (float*)d_ws;
  float* dinv = ws;                             // N
  float* h1   = ws + 50048;                     // N * 64
  float* a1   = h1 + (long)N * N_HID;           // N * 64 (agg1, then relu'd h)
  float* h2   = a1 + (long)N * N_HID;           // N * 32

  const int B = 256;
  const int mtiles = N / 16;                    // 3125 waves, one per 16-row slab

  // 1) degree -> dinv
  deg_init<<<(N + B - 1) / B, B, 0, stream>>>(dinv, N);
  deg_scatter<<<(E + B - 1) / B, B, 0, stream>>>(dst, dinv, E);
  deg_to_rsqrt<<<(N + B - 1) / B, B, 0, stream>>>(dinv, N);

  // 2) h1 = x @ W1   (one wave per 16x64 slab; x streamed once)
  gemm_bf16_wmma<N_FEAT_IN, N_HID><<<(mtiles + 7) / 8, B, 0, stream>>>(x, W1, h1, N);

  // 3) aggregate layer 1: self-loop seed + edge atomics, then bias+relu
  selfloop_init<N_HID><<<((long)N * N_HID + B - 1) / B, B, 0, stream>>>(h1, dinv, a1, N);
  edge_agg<N_HID><<<((long)E * N_HID + B - 1) / B, B, 0, stream>>>(src, dst, h1, dinv, a1, E);
  bias_relu<N_HID><<<((long)N * N_HID + B - 1) / B, B, 0, stream>>>(a1, b1, N * N_HID);

  // 4) h2 = h @ W2   (one wave per 16x32 slab)
  gemm_bf16_wmma<N_HID, N_EMB><<<(mtiles + 7) / 8, B, 0, stream>>>(a1, W2, h2, N);

  // 5) aggregate layer 2 into d_out (fused +b2 in the seed)
  selfloop_init_bias<N_EMB><<<((long)N * N_EMB + B - 1) / B, B, 0, stream>>>(h2, dinv, b2, out, N);
  edge_agg<N_EMB><<<((long)E * N_EMB + B - 1) / B, B, 0, stream>>>(src, dst, h2, dinv, out, E);
}
